// EfficientParticleTransformer_23021024707080
// MI455X (gfx1250) — matrix operand order repaired and checked by reading, not compiled, not run on hardware
//
#include <hip/hip_runtime.h>
#include <hip/hip_bf16.h>
#include <math.h>

// ---------------------------------------------------------------------------
// Types
// ---------------------------------------------------------------------------
typedef _Float16 half_t;
typedef __attribute__((ext_vector_type(16))) _Float16 v16h;
typedef __attribute__((ext_vector_type(8)))  _Float16 v8h;
typedef __attribute__((ext_vector_type(4)))  _Float16 v4h;
typedef __attribute__((ext_vector_type(8)))  float    v8f;
typedef __attribute__((ext_vector_type(4)))  int      v4i;

#define DEV_INLINE __device__ __forceinline__

DEV_INLINE float gelu_exact(float x) {
  return 0.5f * x * (1.0f + erff(x * 0.70710678118654752440f));
}

// ---- CDNA5 async global->LDS copy (guarded: falls back to reg pipeline) ----
#if __has_builtin(__builtin_amdgcn_global_load_async_to_lds_b128) && \
    __has_builtin(__builtin_amdgcn_s_wait_asynccnt)
#define HAVE_ASYNC_LDS 1
typedef v4i __attribute__((address_space(1))) as1_v4i;   // global (AS1) int4
typedef v4i __attribute__((address_space(3))) as3_v4i;   // LDS (AS3) int4
DEV_INLINE void async_copy16(const half_t* g, half_t* l) {
  // AS3 pointers are 32-bit; a generic LDS pointer's low 32 bits are the LDS
  // byte offset (ISA aperture rule), so numeric casts produce valid pointers.
  __builtin_amdgcn_global_load_async_to_lds_b128(
      (as1_v4i*)(unsigned long long)g,
      (as3_v4i*)(unsigned int)(unsigned long long)l,
      0, 0);
}
#else
#define HAVE_ASYNC_LDS 0
#endif

// ---------------------------------------------------------------------------
// Generic batched WMMA GEMM:
//   C[z] = act( scale * (A[z] @ op(B[z])) + bias )
// A: f16 [M,K], row m at A + m*lda  (+ zb*Ab + zh*Ah, z = blockIdx.z, zb=z/nh, zh=z%nh)
// B: f16; bkmajor==0: [N,K] rows (C = A @ B^T);  bkmajor==1: [K,N] rows (C = A @ B)
// C: optional f32 (Cf) and/or f16 (Ch); element (m,n) at m*ldc + n + zb*Cb + zh*Chh
// act: 0 = none, 1 = exact GELU
//
// Block: 256 threads = 8 wave32, 2x4 wave grid; each wave owns a 32x32 C tile
// (2x2 WMMA fragments -> 4 v_wmma per k-step; fragments reused across subtiles).
// Block tile 64x128, k-step 32.
// Fast path (interior tile, K%32==0, vectorizable): double-buffered LDS,
// software-pipelined (async-to-LDS when available, else register prefetch),
// one barrier per k-step. Slow path: clamp+select zero-fill staging (no EXEC
// branching). EXEC is full at every WMMA.
// ---------------------------------------------------------------------------
#define BM 64
#define BN 128
#define TK 32
#define LDK (TK + 8)

__global__ __launch_bounds__(256)
void wmma_gemm_kernel(const half_t* __restrict__ A, long lda, long Ab, long Ah,
                      const half_t* __restrict__ Bm, long ldb, long Bb, long Bh,
                      float* __restrict__ Cf, half_t* __restrict__ Ch,
                      long ldc, long Cb, long Chh,
                      const float* __restrict__ bias,
                      int M, int N, int K, int nh, int bkmajor, int act, float scale)
{
  __shared__ __align__(16) half_t As[2][BM][LDK];
  __shared__ __align__(16) half_t Bs[2][BN][LDK];

  const int z  = blockIdx.z;
  const int zb = z / nh;
  const int zh = z % nh;
  A  += (long)zb * Ab + (long)zh * Ah;
  Bm += (long)zb * Bb + (long)zh * Bh;
  const long coff = (long)zb * Cb + (long)zh * Chh;

  const int tid  = threadIdx.x;
  const int lane = tid & 31;
  const int wave = tid >> 5;
  const int wm   = wave >> 2;   // 0..1  -> 32 rows each
  const int wn   = wave & 3;    // 0..3  -> 32 cols each
  const int m0   = blockIdx.y * BM;
  const int n0   = blockIdx.x * BN;

  v8f acc00 = {}, acc01 = {}, acc10 = {}, acc11 = {};

  // staging coordinates (per thread: one 16B A chunk, two 16B B chunks)
  const int ar = tid >> 2;            // 0..63
  const int ac = (tid & 3) * 8;       // 0,8,16,24

  const bool fast =
      ((lda & 7) == 0) && ((ldb & 7) == 0) && !bkmajor &&
      ((K & (TK - 1)) == 0) && (m0 + BM <= M) && (n0 + BN <= N);

  const int khalf = lane >> 4;
  const int l15   = lane & 15;

  if (fast) {
    const half_t* gA  = A  + (long)(m0 + ar) * lda + ac;
    const half_t* gB0 = Bm + (long)(n0 + ar) * ldb + ac;
    const half_t* gB1 = Bm + (long)(n0 + 64 + ar) * ldb + ac;
    const int nk = K / TK;

#if HAVE_ASYNC_LDS
    // prologue: async stage chunk 0 into buffer 0
    async_copy16(gA,  &As[0][ar][ac]);
    async_copy16(gB0, &Bs[0][ar][ac]);
    async_copy16(gB1, &Bs[0][ar + 64][ac]);
    __builtin_amdgcn_s_wait_asynccnt(0);
    __syncthreads();
    for (int i = 0; i < nk; ++i) {
      const int cur = i & 1, nxt = cur ^ 1;
      if (i + 1 < nk) {   // issue async stage of next chunk; overlap with WMMA
        const int kn = (i + 1) * TK;
        async_copy16(gA  + kn, &As[nxt][ar][ac]);
        async_copy16(gB0 + kn, &Bs[nxt][ar][ac]);
        async_copy16(gB1 + kn, &Bs[nxt][ar + 64][ac]);
      }
      const v8h* arow0 = (const v8h*)&As[cur][wm * 32 + l15][0];
      const v8h* arow1 = (const v8h*)&As[cur][wm * 32 + 16 + l15][0];
      const v8h* brow0 = (const v8h*)&Bs[cur][wn * 32 + l15][0];
      const v8h* brow1 = (const v8h*)&Bs[cur][wn * 32 + 16 + l15][0];
      const v8h a0lo = arow0[khalf],     a0hi = arow0[khalf + 2];
      const v8h a1lo = arow1[khalf],     a1hi = arow1[khalf + 2];
      const v8h b0lo = brow0[khalf],     b0hi = brow0[khalf + 2];
      const v8h b1lo = brow1[khalf],     b1hi = brow1[khalf + 2];
      v16h a0, a1, b0, b1;
      #pragma unroll
      for (int j = 0; j < 8; ++j) {
        a0[j] = a0lo[j]; a0[j + 8] = a0hi[j];
        a1[j] = a1lo[j]; a1[j + 8] = a1hi[j];
        b0[j] = b0lo[j]; b0[j + 8] = b0hi[j];
        b1[j] = b1lo[j]; b1[j + 8] = b1hi[j];
      }
      acc00 = __builtin_amdgcn_wmma_f32_16x16x32_f16(false, a0, false, b0, (short)0, acc00, false, false);
      acc01 = __builtin_amdgcn_wmma_f32_16x16x32_f16(false, a0, false, b1, (short)0, acc01, false, false);
      acc10 = __builtin_amdgcn_wmma_f32_16x16x32_f16(false, a1, false, b0, (short)0, acc10, false, false);
      acc11 = __builtin_amdgcn_wmma_f32_16x16x32_f16(false, a1, false, b1, (short)0, acc11, false, false);
      if (i + 1 < nk) __builtin_amdgcn_s_wait_asynccnt(0);
      __syncthreads();
    }
#else
    // prologue: register-load chunk 0, store to buffer 0
    v8h ra  = *(const v8h*)gA;
    v8h rb0 = *(const v8h*)gB0;
    v8h rb1 = *(const v8h*)gB1;
    *(v8h*)&As[0][ar][ac]      = ra;
    *(v8h*)&Bs[0][ar][ac]      = rb0;
    *(v8h*)&Bs[0][ar + 64][ac] = rb1;
    __syncthreads();
    for (int i = 0; i < nk; ++i) {
      const int cur = i & 1, nxt = cur ^ 1;
      if (i + 1 < nk) {   // prefetch next chunk into registers (overlaps WMMA)
        const int kn = (i + 1) * TK;
        ra  = *(const v8h*)(gA  + kn);
        rb0 = *(const v8h*)(gB0 + kn);
        rb1 = *(const v8h*)(gB1 + kn);
      }
      const v8h* arow0 = (const v8h*)&As[cur][wm * 32 + l15][0];
      const v8h* arow1 = (const v8h*)&As[cur][wm * 32 + 16 + l15][0];
      const v8h* brow0 = (const v8h*)&Bs[cur][wn * 32 + l15][0];
      const v8h* brow1 = (const v8h*)&Bs[cur][wn * 32 + 16 + l15][0];
      const v8h a0lo = arow0[khalf],     a0hi = arow0[khalf + 2];
      const v8h a1lo = arow1[khalf],     a1hi = arow1[khalf + 2];
      const v8h b0lo = brow0[khalf],     b0hi = brow0[khalf + 2];
      const v8h b1lo = brow1[khalf],     b1hi = brow1[khalf + 2];
      v16h a0, a1, b0, b1;
      #pragma unroll
      for (int j = 0; j < 8; ++j) {
        a0[j] = a0lo[j]; a0[j + 8] = a0hi[j];
        a1[j] = a1lo[j]; a1[j + 8] = a1hi[j];
        b0[j] = b0lo[j]; b0[j + 8] = b0hi[j];
        b1[j] = b1lo[j]; b1[j + 8] = b1hi[j];
      }
      acc00 = __builtin_amdgcn_wmma_f32_16x16x32_f16(false, a0, false, b0, (short)0, acc00, false, false);
      acc01 = __builtin_amdgcn_wmma_f32_16x16x32_f16(false, a0, false, b1, (short)0, acc01, false, false);
      acc10 = __builtin_amdgcn_wmma_f32_16x16x32_f16(false, a1, false, b0, (short)0, acc10, false, false);
      acc11 = __builtin_amdgcn_wmma_f32_16x16x32_f16(false, a1, false, b1, (short)0, acc11, false, false);
      if (i + 1 < nk) {
        *(v8h*)&As[nxt][ar][ac]      = ra;
        *(v8h*)&Bs[nxt][ar][ac]      = rb0;
        *(v8h*)&Bs[nxt][ar + 64][ac] = rb1;
      }
      __syncthreads();
    }
#endif
  } else {
    // ---- generic guarded path: clamp addresses, select-zero OOB ----
    for (int k0 = 0; k0 < K; k0 += TK) {
      {
        const int gr  = m0 + ar;
        const long grc = (gr < M ? gr : M - 1);
        #pragma unroll
        for (int j = 0; j < 8; ++j) {
          const int kk  = k0 + ac + j;
          const int kkc = (kk < K ? kk : K - 1);
          const half_t vv = A[grc * lda + kkc];
          As[0][ar][ac + j] = (gr < M && kk < K) ? vv : (half_t)0.f;
        }
      }
      #pragma unroll
      for (int hh = 0; hh < 2; ++hh) {
        const int r   = ar + hh * 64;
        const int gn  = n0 + r;
        const long gnc = (gn < N ? gn : N - 1);
        #pragma unroll
        for (int j = 0; j < 8; ++j) {
          const int kk  = k0 + ac + j;
          const long kkc = (kk < K ? kk : K - 1);
          const half_t vv = bkmajor ? Bm[kkc * ldb + gnc] : Bm[gnc * ldb + kkc];
          Bs[0][r][ac + j] = (gn < N && kk < K) ? vv : (half_t)0.f;
        }
      }
      __syncthreads();

      const v8h* arow0 = (const v8h*)&As[0][wm * 32 + l15][0];
      const v8h* arow1 = (const v8h*)&As[0][wm * 32 + 16 + l15][0];
      const v8h* brow0 = (const v8h*)&Bs[0][wn * 32 + l15][0];
      const v8h* brow1 = (const v8h*)&Bs[0][wn * 32 + 16 + l15][0];
      const v8h a0lo = arow0[khalf],     a0hi = arow0[khalf + 2];
      const v8h a1lo = arow1[khalf],     a1hi = arow1[khalf + 2];
      const v8h b0lo = brow0[khalf],     b0hi = brow0[khalf + 2];
      const v8h b1lo = brow1[khalf],     b1hi = brow1[khalf + 2];
      v16h a0, a1, b0, b1;
      #pragma unroll
      for (int j = 0; j < 8; ++j) {
        a0[j] = a0lo[j]; a0[j + 8] = a0hi[j];
        a1[j] = a1lo[j]; a1[j + 8] = a1hi[j];
        b0[j] = b0lo[j]; b0[j + 8] = b0hi[j];
        b1[j] = b1lo[j]; b1[j + 8] = b1hi[j];
      }
      acc00 = __builtin_amdgcn_wmma_f32_16x16x32_f16(false, a0, false, b0, (short)0, acc00, false, false);
      acc01 = __builtin_amdgcn_wmma_f32_16x16x32_f16(false, a0, false, b1, (short)0, acc01, false, false);
      acc10 = __builtin_amdgcn_wmma_f32_16x16x32_f16(false, a1, false, b0, (short)0, acc10, false, false);
      acc11 = __builtin_amdgcn_wmma_f32_16x16x32_f16(false, a1, false, b1, (short)0, acc11, false, false);
      __syncthreads();
    }
  }

  // ---- epilogue: C layout lane n = lane&15, VGPR r -> M = r (+8 for hi lanes) ----
  const int mbase = m0 + wm * 32 + ((lane >> 4) << 3);
  const int nbase = n0 + wn * 32 + l15;
  const v8f* accs[4] = { &acc00, &acc01, &acc10, &acc11 };
  #pragma unroll
  for (int ti = 0; ti < 2; ++ti) {
    #pragma unroll
    for (int tj = 0; tj < 2; ++tj) {
      const v8f& a = *accs[ti * 2 + tj];
      const int n = nbase + tj * 16;
      #pragma unroll
      for (int r = 0; r < 8; ++r) {
        const int m = mbase + ti * 16 + r;
        if (m < M && n < N) {
          float val = a[r] * scale;
          if (bias) val += bias[n];
          if (act == 1) val = gelu_exact(val);
          const long idx = (long)m * ldc + n + coff;
          if (Cf) Cf[idx] = val;
          if (Ch) Ch[idx] = (half_t)val;
        }
      }
    }
  }
}

// ---------------------------------------------------------------------------
// LayerNorm over last dim D: one block (256 thr) per row; f32 in, f32/f16 out
// ---------------------------------------------------------------------------
__global__ __launch_bounds__(256)
void layernorm_kernel(const float* __restrict__ x, const float* __restrict__ g,
                      const float* __restrict__ bta, float* __restrict__ outf,
                      half_t* __restrict__ outh, long R, int Dd)
{
  __shared__ float red[16];
  const long row = blockIdx.x;
  if (row >= R) return;
  const float* xr = x + row * Dd;
  float s = 0.f, s2 = 0.f;
  for (int d = threadIdx.x; d < Dd; d += blockDim.x) {
    const float v = xr[d]; s += v; s2 += v * v;
  }
  #pragma unroll
  for (int off = 16; off >= 1; off >>= 1) {
    s  += __shfl_xor(s,  off, 32);
    s2 += __shfl_xor(s2, off, 32);
  }
  const int wave = threadIdx.x >> 5, lane = threadIdx.x & 31;
  if (lane == 0) { red[wave] = s; red[8 + wave] = s2; }
  __syncthreads();
  if (wave == 0) {
    float a  = (lane < 8) ? red[lane]     : 0.f;
    float a2 = (lane < 8) ? red[8 + lane] : 0.f;
    #pragma unroll
    for (int off = 4; off >= 1; off >>= 1) {
      a += __shfl_xor(a, off, 32); a2 += __shfl_xor(a2, off, 32);
    }
    if (lane == 0) { red[0] = a; red[1] = a2; }
  }
  __syncthreads();
  const float mu   = red[0] / Dd;
  const float var  = red[1] / Dd - mu * mu;
  const float rstd = rsqrtf(var + 1e-5f);
  for (int d = threadIdx.x; d < Dd; d += blockDim.x) {
    const float v = (xr[d] - mu) * rstd * g[d] + bta[d];
    const long idx = row * Dd + d;
    if (outf) outf[idx] = v;
    if (outh) outh[idx] = (half_t)v;
  }
}

// ---------------------------------------------------------------------------
// Masked / biased softmax. scores f32 [B*H*T, S]; bias f32 [B,T,S,H] (optional);
// kpm bool [B,S] (or [B,S-1] shifted when maskShift, row 0 unmasked = cls slot).
// attn out f16, same row layout as scores.  One wave per row.
// ---------------------------------------------------------------------------
__global__ __launch_bounds__(128)
void softmax_kernel(const float* __restrict__ scores, const float* __restrict__ biasf,
                    const unsigned char* __restrict__ kpm, half_t* __restrict__ attn,
                    int Bn, int Hn, int Tn, int Sn, int maskShift)
{
  const int  lane = threadIdx.x & 31;
  const long row  = (long)blockIdx.x * (blockDim.x >> 5) + (threadIdx.x >> 5);
  const long nrows = (long)Bn * Hn * Tn;
  if (row >= nrows) return;
  const int  t  = (int)(row % Tn);
  const long zz = row / Tn;
  const int  h  = (int)(zz % Hn);
  const int  b  = (int)(zz / Hn);
  const float* sr = scores + row * Sn;

  float vals[8];
  int   cnt = 0;
  float mx  = -3.4e38f;
  for (int s = lane; s < Sn; s += 32) {
    float v = sr[s];
    if (biasf) v += biasf[(((long)b * Tn + t) * Sn + s) * Hn + h];
    bool m;
    if (maskShift) m = (s == 0) ? false : (kpm[(long)b * (Sn - 1) + (s - 1)] != 0);
    else           m = (kpm[(long)b * Sn + s] != 0);
    if (m) v = -1.0e9f;
    if (cnt < 8) vals[cnt] = v;
    ++cnt;
    mx = fmaxf(mx, v);
  }
  #pragma unroll
  for (int off = 16; off >= 1; off >>= 1) mx = fmaxf(mx, __shfl_xor(mx, off, 32));
  float sum = 0.f;
  for (int i = 0; i < cnt; ++i) { vals[i] = __expf(vals[i] - mx); sum += vals[i]; }
  #pragma unroll
  for (int off = 16; off >= 1; off >>= 1) sum += __shfl_xor(sum, off, 32);
  const float inv = 1.f / sum;
  int i = 0;
  for (int s = lane; s < Sn; s += 32) { attn[row * Sn + s] = (half_t)(vals[i] * inv); ++i; }
}

// ---------------------------------------------------------------------------
// Elementwise helpers
// ---------------------------------------------------------------------------
__global__ void cvt_f32_to_f16_kernel(const float* __restrict__ src,
                                      half_t* __restrict__ dst, long n)
{
  const long i = (long)blockIdx.x * blockDim.x + threadIdx.x;
  if (i < n) dst[i] = (half_t)src[i];
}

// h[n*B + b, c] = x[b, c, n]
__global__ void transpose_in_kernel(const float* __restrict__ x, float* __restrict__ h,
                                    int Bn, int Cn, int Nn)
{
  const long i = (long)blockIdx.x * blockDim.x + threadIdx.x;
  const long total = (long)Nn * Bn * Cn;
  if (i >= total) return;
  const int  c = (int)(i % Cn);
  const long r = i / Cn;
  const int  b = (int)(r % Bn);
  const int  n = (int)(r / Bn);
  h[i] = x[((long)b * Cn + c) * Nn + n];
}

// out[r, d*H + h] = a[r, h*Dh + d] * c_attn[h]
__global__ void head_mix_kernel(const float* __restrict__ a, const float* __restrict__ ca,
                                float* __restrict__ out, long rows, int Hn, int Dhn)
{
  const long i = (long)blockIdx.x * blockDim.x + threadIdx.x;
  const int  Dd = Hn * Dhn;
  const long total = rows * Dd;
  if (i >= total) return;
  const long r = i / Dd;
  const int  c = (int)(i % Dd);
  const int  d = c / Hn, h = c % Hn;
  out[i] = a[r * Dd + (long)h * Dhn + d] * ca[h];
}

__global__ void add_kernel(const float* __restrict__ a, const float* __restrict__ b,
                           float* __restrict__ o, long n)
{
  const long i = (long)blockIdx.x * blockDim.x + threadIdx.x;
  if (i < n) o[i] = a[i] + b[i];
}

// o[r,d] = w[d]*res[r,d] + u[r,d]
__global__ void wresid_kernel(const float* __restrict__ w, const float* __restrict__ res,
                              const float* __restrict__ u, float* __restrict__ o,
                              long rows, int Dd)
{
  const long i = (long)blockIdx.x * blockDim.x + threadIdx.x;
  const long total = rows * Dd;
  if (i >= total) return;
  o[i] = w[i % Dd] * res[i] + u[i];
}

__global__ void broadcast_cls_kernel(const float* __restrict__ tok,
                                     float* __restrict__ cls, int Bn, int Dd)
{
  const long i = (long)blockIdx.x * blockDim.x + threadIdx.x;
  if (i < (long)Bn * Dd) cls[i] = tok[i % Dd];
}

// u[(s*B + b), d] = s==0 ? cls[b,d] : h[((s-1)*B + b), d]
__global__ void concat_cls_kernel(const float* __restrict__ cls, const float* __restrict__ h,
                                  float* __restrict__ u, int Nn, int Bn, int Dd)
{
  const long i = (long)blockIdx.x * blockDim.x + threadIdx.x;
  const long total = (long)(Nn + 1) * Bn * Dd;
  if (i >= total) return;
  const int  d = (int)(i % Dd);
  const long r = i / Dd;
  const int  b = (int)(r % Bn);
  const int  s = (int)(r / Bn);
  u[i] = (s == 0) ? cls[(long)b * Dd + d] : h[((long)(s - 1) * Bn + b) * Dd + d];
}

// Pairwise kinematic features -> f16 [B*N*N, 4] = (lnkt, lnz, lndelta, lnm2)
__global__ void pair_features_kernel(const float* __restrict__ v, half_t* __restrict__ f,
                                     int Bn, int Nn)
{
  const long idx = (long)blockIdx.x * blockDim.x + threadIdx.x;
  const long total = (long)Bn * Nn * Nn;
  if (idx >= total) return;
  const int  j  = (int)(idx % Nn);
  const long tt = idx / Nn;
  const int  i  = (int)(tt % Nn);
  const int  b  = (int)(tt / Nn);
  const float* vb = v + (long)b * 4 * Nn;
  const float pxi = vb[i],          pyi = vb[Nn + i];
  const float pzi = vb[2 * Nn + i], Ei  = vb[3 * Nn + i];
  const float pxj = vb[j],          pyj = vb[Nn + j];
  const float pzj = vb[2 * Nn + j], Ej  = vb[3 * Nn + j];
  const float PI2 = 6.283185307179586f;
  const float pti = sqrtf(pxi * pxi + pyi * pyi);
  const float ptj = sqrtf(pxj * pxj + pyj * pyj);
  const float rapi = 0.5f * log1pf(2.f * pzi / fmaxf(Ei - pzi, 1e-20f));
  const float rapj = 0.5f * log1pf(2.f * pzj / fmaxf(Ej - pzj, 1e-20f));
  const float phii = atan2f(pyi, pxi);
  const float phij = atan2f(pyj, pxj);
  const float drap = rapi - rapj;
  float dp = phii - phij + 3.14159265358979f;
  dp = dp - floorf(dp / PI2) * PI2;            // jnp.mod (always-positive)
  const float dphi  = dp - 3.14159265358979f;
  const float delta = sqrtf(drap * drap + dphi * dphi);
  const float lnd   = logf(fmaxf(delta, 1e-8f));
  const float ptmin = fminf(pti, ptj);
  const float lnkt  = logf(fmaxf(ptmin * delta, 1e-8f));
  const float lnz   = logf(fmaxf(ptmin / fmaxf(pti + ptj, 1e-8f), 1e-8f));
  const float Es = Ei + Ej;
  const float sx = pxi + pxj, sy = pyi + pyj, sz = pzi + pzj;
  const float lnm2 = logf(fmaxf(Es * Es - (sx * sx + sy * sy + sz * sz), 1e-8f));
  half_t* fr = f + idx * 4;
  fr[0] = (half_t)lnkt; fr[1] = (half_t)lnz; fr[2] = (half_t)lnd; fr[3] = (half_t)lnm2;
}

// ---------------------------------------------------------------------------
// Host orchestration
// ---------------------------------------------------------------------------
namespace cfg {
  constexpr int B = 64, N = 128, C = 17, D = 256, H = 8, Dh = 32,
                F = 1024, L = 8, LC = 2, NC = 10;
}

static inline unsigned cdiv(long a, long b) { return (unsigned)((a + b - 1) / b); }

struct Bump {
  char* base; size_t off, cap;
  void* take(size_t bytes) {
    const size_t a = (bytes + 255) & ~(size_t)255;
    void* r = base + off; off += a; return r;
  }
};

static void gemm(hipStream_t s,
                 const half_t* A, long lda, long Ab, long Ah,
                 const half_t* Bm, long ldb, long Bb, long Bh,
                 float* Cf, half_t* Ch, long ldc, long Cb, long Chh,
                 const float* bias, int M, int N, int K,
                 int batches, int nh, int bkmajor, int act, float scale)
{
  dim3 grid(cdiv(N, BN), cdiv(M, BM), (unsigned)batches);
  wmma_gemm_kernel<<<grid, 256, 0, s>>>(A, lda, Ab, Ah, Bm, ldb, Bb, Bh,
                                        Cf, Ch, ldc, Cb, Chh, bias,
                                        M, N, K, nh, bkmajor, act, scale);
}

static half_t* to_h(hipStream_t s, Bump& w, const float* src, long n)
{
  half_t* dst = (half_t*)w.take((size_t)n * sizeof(half_t));
  cvt_f32_to_f16_kernel<<<cdiv(n, 256), 256, 0, s>>>(src, dst, n);
  return dst;
}

// Param flattening (jax pytree: dict keys sorted, lists in order) after x, v, mask.
// Block leaves sorted: W1 W2 Wo Wqkv b1 b2 bo bqkv c_attn ln1_b ln1_g ln2_b ln2_g
//                      ln3_b ln3_g ln4_b ln4_g w_resid
enum {
  L_W1 = 0, L_W2, L_Wo, L_Wqkv, L_b1, L_b2, L_bo, L_bqkv, L_c_attn,
  L_ln1_b, L_ln1_g, L_ln2_b, L_ln2_g, L_ln3_b, L_ln3_g, L_ln4_b, L_ln4_g, L_w_resid
};
static const float* BP(void* const* din, int blk, int leaf, bool isCls)
{
  const int base = 3 + (isCls ? cfg::L * 18 : 0) + blk * 18;
  return (const float*)din[base + leaf];
}
// top-level: blocks(144) cls_blocks(36) cls_token(1) embed(12) fc_W fc_b norm_b norm_g pair(6)
static const int I_CLS_TOKEN = 3 + 10 * 18;           // 183
static const int I_EMBED     = I_CLS_TOKEN + 1;       // 184 (+ i*4 + {W,b,ln_b,ln_g})
static const int I_FCW       = I_EMBED + 12;          // 196
static const int I_FCB       = I_FCW + 1;             // 197
static const int I_NORMB     = I_FCB + 1;             // 198
static const int I_NORMG     = I_NORMB + 1;           // 199
static const int I_PAIR      = I_NORMG + 1;           // 200 (W1 W2 W3 b1 b2 b3)

extern "C" void kernel_launch(void* const* d_in, const int* in_sizes, int n_in,
                              void* d_out, int out_size, void* d_ws, size_t ws_size,
                              hipStream_t stream)
{
  using namespace cfg;
  (void)in_sizes; (void)n_in; (void)out_size;

  const float*         x    = (const float*)d_in[0];
  const float*         vkin = (const float*)d_in[1];
  const unsigned char* kpm  = (const unsigned char*)d_in[2];

  Bump w{ (char*)d_ws, 0, ws_size };
  const long RB = (long)N * B;          // 8192 token rows
  const int  S2 = N + 1;                // cls attention length

  // ---- weights -> f16 ----
  half_t *Wqkv_h[L + LC], *Wo_h[L + LC], *W1_h[L + LC], *W2_h[L + LC];
  for (int l = 0; l < L + LC; ++l) {
    const bool c = l >= L; const int bi = c ? l - L : l;
    Wqkv_h[l] = to_h(stream, w, BP(d_in, bi, L_Wqkv, c), 3L * D * D);
    Wo_h[l]   = to_h(stream, w, BP(d_in, bi, L_Wo,   c), (long)D * D);
    W1_h[l]   = to_h(stream, w, BP(d_in, bi, L_W1,   c), (long)F * D);
    W2_h[l]   = to_h(stream, w, BP(d_in, bi, L_W2,   c), (long)D * F);
  }
  const int edims[4] = { C, 128, 512, D };
  half_t* eW_h[3];
  for (int i = 0; i < 3; ++i)
    eW_h[i] = to_h(stream, w, (const float*)d_in[I_EMBED + i * 4 + 0],
                   (long)edims[i + 1] * edims[i]);
  half_t* pW1_h = to_h(stream, w, (const float*)d_in[I_PAIR + 0], 64L * 4);
  half_t* pW2_h = to_h(stream, w, (const float*)d_in[I_PAIR + 1], 64L * 64);
  half_t* pW3_h = to_h(stream, w, (const float*)d_in[I_PAIR + 2], (long)H * 64);
  half_t* fcW_h = to_h(stream, w, (const float*)d_in[I_FCW], (long)NC * D);

  // ---- activations ----
  float*  ebufA  = (float*)w.take(RB * 512 * 4);
  float*  ebufB  = (float*)w.take(RB * 512 * 4);
  float*  hbuf   = (float*)w.take(RB * D * 4);
  half_t* u_h    = (half_t*)w.take(RB * F * 2);
  half_t* qkv_h  = (half_t*)w.take(RB * 3 * D * 2);
  float*  scores = (float*)w.take((long)B * H * N * N * 4);
  half_t* attn_h = (half_t*)w.take((long)B * H * N * N * 2);
  half_t* ao_h   = (half_t*)w.take(RB * D * 2);
  float*  a_f    = (float*)w.take(RB * D * 4);
  float*  am_f   = (float*)w.take(RB * D * 4);
  float*  a2_f   = (float*)w.take(RB * D * 4);
  float*  h2_f   = (float*)w.take(RB * D * 4);
  float*  u4_f   = (float*)w.take(RB * F * 4);
  float*  u6_f   = (float*)w.take(RB * D * 4);
  float*  pairf3 = (float*)w.take((long)B * N * N * H * 4);
  half_t* pfeat  = (half_t*)w.take((long)B * N * N * 4 * 2);
  const int CH = 4;
  half_t* phid1  = (half_t*)w.take((long)CH * N * N * 64 * 2);
  half_t* phid2  = (half_t*)w.take((long)CH * N * N * 64 * 2);
  float*  cls_f  = (float*)w.take((long)B * D * 4);
  half_t* cls_h  = (half_t*)w.take((long)B * D * 2);
  float*  ucat   = (float*)w.take((long)S2 * B * D * 4);
  half_t* kv_h   = (half_t*)w.take((long)S2 * B * 2 * D * 2);
  half_t* cq_h   = (half_t*)w.take((long)B * D * 2);
  float*  csc    = (float*)w.take((long)B * H * S2 * 4);
  half_t* catt   = (half_t*)w.take((long)B * H * S2 * 2);

  // ---- embedding MLP: h = transpose(x) ; 3x (LN -> Linear -> GELU) ----
  transpose_in_kernel<<<cdiv(RB * C, 256), 256, 0, stream>>>(x, ebufA, B, C, N);
  {
    float* cur = ebufA;
    for (int i = 0; i < 3; ++i) {
      const float* lg = (const float*)d_in[I_EMBED + i * 4 + 3];  // ln_g
      const float* lb = (const float*)d_in[I_EMBED + i * 4 + 2];  // ln_b
      const float* bb = (const float*)d_in[I_EMBED + i * 4 + 1];  // b
      layernorm_kernel<<<(unsigned)RB, 256, 0, stream>>>(cur, lg, lb, nullptr, u_h, RB, edims[i]);
      float* nxt = (i == 2) ? hbuf : ((cur == ebufA) ? ebufB : ebufA);
      gemm(stream, u_h, edims[i], 0, 0, eW_h[i], edims[i], 0, 0,
           nxt, nullptr, edims[i + 1], 0, 0, bb,
           (int)RB, edims[i + 1], edims[i], 1, 1, 0, /*gelu*/1, 1.f);
      cur = nxt;
    }
  }

  // ---- pairwise bias: features -> 4->64->64->8 MLP (chunked over batch) ----
  pair_features_kernel<<<cdiv((long)B * N * N, 256), 256, 0, stream>>>(vkin, pfeat, B, N);
  for (int c0 = 0; c0 < B; c0 += CH) {
    const long Mc = (long)CH * N * N;
    gemm(stream, pfeat + (long)c0 * N * N * 4, 4, 0, 0, pW1_h, 4, 0, 0,
         nullptr, phid1, 64, 0, 0, (const float*)d_in[I_PAIR + 3],
         (int)Mc, 64, 4, 1, 1, 0, 1, 1.f);
    gemm(stream, phid1, 64, 0, 0, pW2_h, 64, 0, 0,
         nullptr, phid2, 64, 0, 0, (const float*)d_in[I_PAIR + 4],
         (int)Mc, 64, 64, 1, 1, 0, 1, 1.f);
    gemm(stream, phid2, 64, 0, 0, pW3_h, 64, 0, 0,
         pairf3 + (long)c0 * N * N * H, nullptr, H, 0, 0, (const float*)d_in[I_PAIR + 5],
         (int)Mc, H, 64, 1, 1, 0, 0, 1.f);
  }

  const float iss = 0.17677669529663687f;  // 1/sqrt(Dh)

  // ---- main transformer blocks ----
  for (int l = 0; l < L; ++l) {
    layernorm_kernel<<<(unsigned)RB, 256, 0, stream>>>(
        hbuf, BP(d_in, l, L_ln1_g, false), BP(d_in, l, L_ln1_b, false),
        nullptr, u_h, RB, D);
    gemm(stream, u_h, D, 0, 0, Wqkv_h[l], D, 0, 0,
         nullptr, qkv_h, 3 * D, 0, 0, BP(d_in, l, L_bqkv, false),
         (int)RB, 3 * D, D, 1, 1, 0, 0, 1.f);
    // scores[b,h,t,s] = q.k / sqrt(Dh)
    gemm(stream, qkv_h, (long)B * 3 * D, 3 * D, Dh,
         qkv_h + D, (long)B * 3 * D, 3 * D, Dh,
         scores, nullptr, N, (long)H * N * N, (long)N * N,
         nullptr, N, N, Dh, B * H, H, 0, 0, iss);
    softmax_kernel<<<cdiv((long)B * H * N, 4), 128, 0, stream>>>(
        scores, pairf3, kpm, attn_h, B, H, N, N, 0);
    // out[t,b,h,d] = attn @ v
    gemm(stream, attn_h, N, (long)H * N * N, (long)N * N,
         qkv_h + 2 * D, (long)B * 3 * D, 3 * D, Dh,
         nullptr, ao_h, (long)B * D, D, Dh,
         nullptr, N, Dh, N, B * H, H, /*bkmajor*/1, 0, 1.f);
    gemm(stream, ao_h, D, 0, 0, Wo_h[l], D, 0, 0,
         a_f, nullptr, D, 0, 0, BP(d_in, l, L_bo, false),
         (int)RB, D, D, 1, 1, 0, 0, 1.f);
    head_mix_kernel<<<cdiv(RB * D, 256), 256, 0, stream>>>(
        a_f, BP(d_in, l, L_c_attn, false), am_f, RB, H, Dh);
    layernorm_kernel<<<(unsigned)RB, 256, 0, stream>>>(
        am_f, BP(d_in, l, L_ln2_g, false), BP(d_in, l, L_ln2_b, false),
        a2_f, nullptr, RB, D);
    add_kernel<<<cdiv(RB * D, 256), 256, 0, stream>>>(hbuf, a2_f, h2_f, RB * D);
    layernorm_kernel<<<(unsigned)RB, 256, 0, stream>>>(
        h2_f, BP(d_in, l, L_ln3_g, false), BP(d_in, l, L_ln3_b, false),
        nullptr, u_h, RB, D);
    gemm(stream, u_h, D, 0, 0, W1_h[l], D, 0, 0,
         u4_f, nullptr, F, 0, 0, BP(d_in, l, L_b1, false),
         (int)RB, F, D, 1, 1, 0, 1, 1.f);
    layernorm_kernel<<<(unsigned)RB, 256, 0, stream>>>(
        u4_f, BP(d_in, l, L_ln4_g, false), BP(d_in, l, L_ln4_b, false),
        nullptr, u_h, RB, F);
    gemm(stream, u_h, F, 0, 0, W2_h[l], F, 0, 0,
         u6_f, nullptr, D, 0, 0, BP(d_in, l, L_b2, false),
         (int)RB, D, F, 1, 1, 0, 0, 1.f);
    wresid_kernel<<<cdiv(RB * D, 256), 256, 0, stream>>>(
        BP(d_in, l, L_w_resid, false), h2_f, u6_f, hbuf, RB, D);
  }

  // ---- class-attention blocks ----
  broadcast_cls_kernel<<<cdiv((long)B * D, 256), 256, 0, stream>>>(
      (const float*)d_in[I_CLS_TOKEN], cls_f, B, D);
  for (int lc = 0; lc < LC; ++lc) {
    const int l = L + lc;
    const long RU = (long)S2 * B;
    concat_cls_kernel<<<cdiv(RU * D, 256), 256, 0, stream>>>(cls_f, hbuf, ucat, N, B, D);
    layernorm_kernel<<<(unsigned)RU, 256, 0, stream>>>(
        ucat, BP(d_in, lc, L_ln1_g, true), BP(d_in, lc, L_ln1_b, true),
        nullptr, u_h, RU, D);
    // k,v from u
    gemm(stream, u_h, D, 0, 0, Wqkv_h[l] + (long)D * D, D, 0, 0,
         nullptr, kv_h, 2 * D, 0, 0, BP(d_in, lc, L_bqkv, true) + D,
         (int)RU, 2 * D, D, 1, 1, 0, 0, 1.f);
    // q from raw x_cls
    cvt_f32_to_f16_kernel<<<cdiv((long)B * D, 256), 256, 0, stream>>>(cls_f, cls_h, (long)B * D);
    gemm(stream, cls_h, D, 0, 0, Wqkv_h[l], D, 0, 0,
         nullptr, cq_h, D, 0, 0, BP(d_in, lc, L_bqkv, true),
         B, D, D, 1, 1, 0, 0, 1.f);
    gemm(stream, cq_h, D, D, Dh,
         kv_h, (long)B * 2 * D, 2 * D, Dh,
         csc, nullptr, S2, (long)H * S2, S2,
         nullptr, 1, S2, Dh, B * H, H, 0, 0, iss);
    softmax_kernel<<<cdiv((long)B * H, 4), 128, 0, stream>>>(
        csc, nullptr, kpm, catt, B, H, 1, S2, /*maskShift*/1);
    gemm(stream, catt, S2, (long)H * S2, S2,
         kv_h + D, (long)B * 2 * D, 2 * D, Dh,
         nullptr, ao_h, (long)B * D, D, Dh,
         nullptr, 1, Dh, S2, B * H, H, 1, 0, 1.f);
    gemm(stream, ao_h, D, 0, 0, Wo_h[l], D, 0, 0,
         a_f, nullptr, D, 0, 0, BP(d_in, lc, L_bo, true),
         B, D, D, 1, 1, 0, 0, 1.f);
    head_mix_kernel<<<cdiv((long)B * D, 256), 256, 0, stream>>>(
        a_f, BP(d_in, lc, L_c_attn, true), am_f, B, H, Dh);
    layernorm_kernel<<<(unsigned)B, 256, 0, stream>>>(
        am_f, BP(d_in, lc, L_ln2_g, true), BP(d_in, lc, L_ln2_b, true),
        a2_f, nullptr, B, D);
    add_kernel<<<cdiv((long)B * D, 256), 256, 0, stream>>>(cls_f, a2_f, h2_f, (long)B * D);
    layernorm_kernel<<<(unsigned)B, 256, 0, stream>>>(
        h2_f, BP(d_in, lc, L_ln3_g, true), BP(d_in, lc, L_ln3_b, true),
        nullptr, u_h, B, D);
    gemm(stream, u_h, D, 0, 0, W1_h[l], D, 0, 0,
         u4_f, nullptr, F, 0, 0, BP(d_in, lc, L_b1, true),
         B, F, D, 1, 1, 0, 1, 1.f);
    layernorm_kernel<<<(unsigned)B, 256, 0, stream>>>(
        u4_f, BP(d_in, lc, L_ln4_g, true), BP(d_in, lc, L_ln4_b, true),
        nullptr, u_h, B, F);
    gemm(stream, u_h, F, 0, 0, W2_h[l], F, 0, 0,
         u6_f, nullptr, D, 0, 0, BP(d_in, lc, L_b2, true),
         B, D, F, 1, 1, 0, 0, 1.f);
    wresid_kernel<<<cdiv((long)B * D, 256), 256, 0, stream>>>(
        BP(d_in, lc, L_w_resid, true), h2_f, u6_f, cls_f, B, D);
  }

  // ---- final norm + classifier ----
  layernorm_kernel<<<(unsigned)B, 256, 0, stream>>>(
      cls_f, (const float*)d_in[I_NORMG], (const float*)d_in[I_NORMB],
      nullptr, u_h, B, D);
  gemm(stream, u_h, D, 0, 0, fcW_h, D, 0, 0,
       (float*)d_out, nullptr, NC, 0, 0, (const float*)d_in[I_FCB],
       B, NC, D, 1, 1, 0, 0, 1.f);
}
